// QuantSwiGLU_79182017069419
// MI455X (gfx1250) — matrix operand order repaired and checked by reading, not compiled, not run on hardware
//
#include <hip/hip_runtime.h>
#include <hip/hip_bf16.h>

// ---------------------------------------------------------------------------
// QuantSwiGLU on MI455X (gfx1250): bf16 WMMA + async-to-LDS double buffering.
//   M = 8192 tokens, D = 2048, H = 8192.
// Steady-state loop is branch-free: unrolled x2 with constant buffer index,
// last two K-tiles peeled.
// ---------------------------------------------------------------------------

#define M_TOT 8192
#define D_DIM 2048
#define H_DIM 8192

typedef __attribute__((ext_vector_type(16))) __bf16 v16bf;
typedef __attribute__((ext_vector_type(8)))  __bf16 v8bf;
typedef __attribute__((ext_vector_type(8)))  float  v8f;

union Frag { v16bf v; v8bf h[2]; };

#define LDA 40  // padded LDS row stride (32 + 8) in bf16 elements

// LDS byte offset of a generic pointer into __shared__ (addrspace(3) low bits)
__device__ __forceinline__ unsigned lds_off(const void* p) {
    return (unsigned)(size_t)p;
}

__device__ __forceinline__ void async_b128(unsigned loff, const void* gaddr) {
    asm volatile("global_load_async_to_lds_b128 %0, %1, off"
                 :: "v"(loff), "v"(gaddr) : "memory");
}

__device__ __forceinline__ void wait_async4() {
    asm volatile("s_wait_asynccnt 4" ::: "memory");
}
__device__ __forceinline__ void wait_async0() {
    asm volatile("s_wait_asynccnt 0" ::: "memory");
}

// ---------------------------------------------------------------------------
// Stage 0: zero the absmax cells (ws is poisoned, must re-init every call)
// ---------------------------------------------------------------------------
__global__ void init_cells_kernel(unsigned int* cells) {
    if (threadIdx.x < 3) cells[threadIdx.x] = 0u;
}

// ---------------------------------------------------------------------------
// Stage 1: per-tensor abs-max (atomic umax on float bits; valid for >= 0)
// ---------------------------------------------------------------------------
__global__ __launch_bounds__(256) void absmax_kernel(
    const float* __restrict__ w, size_t n, unsigned int* __restrict__ cell)
{
    __shared__ float red[256];
    float m = 0.0f;
    for (size_t i = (size_t)blockIdx.x * blockDim.x + threadIdx.x; i < n;
         i += (size_t)gridDim.x * blockDim.x)
        m = fmaxf(m, fabsf(w[i]));
    red[threadIdx.x] = m;
    __syncthreads();
    for (int s = 128; s > 0; s >>= 1) {
        if ((int)threadIdx.x < s)
            red[threadIdx.x] = fmaxf(red[threadIdx.x], red[threadIdx.x + s]);
        __syncthreads();
    }
    if (threadIdx.x == 0) atomicMax(cell, __float_as_uint(red[0]));
}

// ---------------------------------------------------------------------------
// Stage 2a: weight fake-quant -> integer-valued bf16 (ints <=127 are exact)
// ---------------------------------------------------------------------------
__global__ __launch_bounds__(256) void quant_weight_kernel(
    const float* __restrict__ w, __bf16* __restrict__ wq, size_t n,
    const unsigned int* __restrict__ cell)
{
    const float s   = __uint_as_float(*cell) * (1.0f / 127.0f);
    const float inv = (s > 0.0f) ? (1.0f / s) : 0.0f;
    for (size_t i = (size_t)blockIdx.x * blockDim.x + threadIdx.x; i < n;
         i += (size_t)gridDim.x * blockDim.x) {
        float r = fminf(fmaxf(rintf(w[i] * inv), -127.0f), 127.0f);
        wq[i] = (__bf16)r;
    }
}

// Stage 2b: activations f32 -> bf16
__global__ __launch_bounds__(256) void cvt_bf16_kernel(
    const float* __restrict__ x, __bf16* __restrict__ xb, size_t n)
{
    for (size_t i = (size_t)blockIdx.x * blockDim.x + threadIdx.x; i < n;
         i += (size_t)gridDim.x * blockDim.x)
        xb[i] = (__bf16)x[i];
}

// ---------------------------------------------------------------------------
// Stage 3: fused GEMM1: gate = x@W1^T, up = x@W3^T, SiLU + act-quant + mul.
// Block 128(M) x 64(N=H), K-step 32, 8 waves, wave tile 32x32 on BOTH mats.
// LDS filled by GLOBAL_LOAD_ASYNC_TO_LDS_B128 (ASYNCcnt), double buffered:
// 4 async instructions per tile per wave -> s_wait_asynccnt 4 overlaps the
// next tile's DMA with this tile's 8 WMMAs.
// ---------------------------------------------------------------------------
__global__ __launch_bounds__(256) void gemm1_swiglu_kernel(
    const __bf16* __restrict__ xb,  const __bf16* __restrict__ w1b,
    const __bf16* __restrict__ w3b, __bf16* __restrict__ hidden,
    const unsigned int* __restrict__ cells,
    const float* __restrict__ sg_p, const float* __restrict__ su_p)
{
    __shared__ __align__(16) __bf16 ldsA [2][128 * LDA];
    __shared__ __align__(16) __bf16 ldsB1[2][ 64 * LDA];
    __shared__ __align__(16) __bf16 ldsB3[2][ 64 * LDA];

    const int m0   = blockIdx.y * 128;
    const int n0   = blockIdx.x * 64;
    const int t    = threadIdx.x;
    const int lane = t & 31;
    const int wv   = t >> 5;
    const int wr   = wv & 3;    // 4 wave-rows  * 32 M
    const int wc   = wv >> 2;   // 2 wave-cols  * 32 N
    const int lrow = lane & 15;
    const int kh   = lane >> 4; // 0/1 lane half

    v8f accG[2][2] = {};
    v8f accU[2][2] = {};

    // Issue one K-tile's DMA: A 128x32 (2 chunks/thread), B1/B3 64x32 (1 each)
    auto issue_tile = [&](int kk, int buf) {
        #pragma unroll
        for (int i = 0; i < 2; ++i) {
            int cc  = t + i * 256;
            int row = cc >> 2, col = (cc & 3) * 8;
            async_b128(lds_off(&ldsA[buf][row * LDA + col]),
                       &xb[(size_t)(m0 + row) * D_DIM + kk + col]);
        }
        int row = t >> 2, col = (t & 3) * 8;
        async_b128(lds_off(&ldsB1[buf][row * LDA + col]),
                   &w1b[(size_t)(n0 + row) * D_DIM + kk + col]);
        async_b128(lds_off(&ldsB3[buf][row * LDA + col]),
                   &w3b[(size_t)(n0 + row) * D_DIM + kk + col]);
    };

    // Consume one K-tile from LDS buffer `buf`: 12 ds_load_b128 -> 8 WMMA
    auto compute_tile = [&](int buf) {
        Frag a[2], b1f[2], b3f[2];
        #pragma unroll
        for (int mi = 0; mi < 2; ++mi) {
            int r = wr * 32 + mi * 16 + lrow;
            a[mi].h[0] = *(const v8bf*)&ldsA[buf][r * LDA + kh * 16];
            a[mi].h[1] = *(const v8bf*)&ldsA[buf][r * LDA + kh * 16 + 8];
        }
        #pragma unroll
        for (int ni = 0; ni < 2; ++ni) {
            int r = wc * 32 + ni * 16 + lrow;
            b1f[ni].h[0] = *(const v8bf*)&ldsB1[buf][r * LDA + kh * 16];
            b1f[ni].h[1] = *(const v8bf*)&ldsB1[buf][r * LDA + kh * 16 + 8];
            b3f[ni].h[0] = *(const v8bf*)&ldsB3[buf][r * LDA + kh * 16];
            b3f[ni].h[1] = *(const v8bf*)&ldsB3[buf][r * LDA + kh * 16 + 8];
        }
        #pragma unroll
        for (int mi = 0; mi < 2; ++mi)
            #pragma unroll
            for (int ni = 0; ni < 2; ++ni) {
                accG[mi][ni] = __builtin_amdgcn_wmma_f32_16x16x32_bf16(
                    false, a[mi].v, false, b1f[ni].v, (short)0,
                    accG[mi][ni], false, false);
                accU[mi][ni] = __builtin_amdgcn_wmma_f32_16x16x32_bf16(
                    false, a[mi].v, false, b3f[ni].v, (short)0,
                    accU[mi][ni], false, false);
            }
    };

    issue_tile(0, 0);

    // Steady state: branch-free, constant buffer index per call site.
    for (int kk = 0; kk < D_DIM - 64; kk += 64) {
        issue_tile(kk + 32, 1);
        wait_async4();
        __syncthreads();
        compute_tile(0);
        __syncthreads();

        issue_tile(kk + 64, 0);
        wait_async4();
        __syncthreads();
        compute_tile(1);
        __syncthreads();
    }
    // Tail: tiles D-64 (buf0, already issued) and D-32 (buf1).
    issue_tile(D_DIM - 32, 1);
    wait_async4();
    __syncthreads();
    compute_tile(0);
    __syncthreads();
    wait_async0();
    __syncthreads();
    compute_tile(1);

    // ---- epilogue: scale, SiLU, act fake-quant, elementwise product ----
    const float s1  = __uint_as_float(cells[0]) * (1.0f / 127.0f);
    const float s3  = __uint_as_float(cells[1]) * (1.0f / 127.0f);
    const float sg  = sg_p[0], su = su_p[0];
    const float isg = 1.0f / sg, isu = 1.0f / su;

    #pragma unroll
    for (int mi = 0; mi < 2; ++mi)
        #pragma unroll
        for (int ni = 0; ni < 2; ++ni) {
            int baser = m0 + wr * 32 + mi * 16 + kh * 8;
            int col   = n0 + wc * 32 + ni * 16 + lrow;
            #pragma unroll
            for (int r = 0; r < 8; ++r) {
                float g = accG[mi][ni][r] * s1;
                g = g / (1.0f + __expf(-g));                         // SiLU
                float gq = fminf(fmaxf(rintf(g * isg), -128.0f), 127.0f) * sg;
                float u  = accU[mi][ni][r] * s3;
                float uq = fminf(fmaxf(rintf(u * isu), -128.0f), 127.0f) * su;
                hidden[(size_t)(baser + r) * H_DIM + col] = (__bf16)(gq * uq);
            }
        }
}

// ---------------------------------------------------------------------------
// Stage 4: GEMM2: out = fake_quant_act(hidden @ W2^T * s2, s_out), f32 out.
// Block 128(M) x 128(N=D), wave tile 32x64 (2x4 WMMA), K = H = 8192.
// Same async double-buffered, branch-free pipeline.
// ---------------------------------------------------------------------------
__global__ __launch_bounds__(256) void gemm2_kernel(
    const __bf16* __restrict__ hid, const __bf16* __restrict__ w2b,
    float* __restrict__ out, const unsigned int* __restrict__ cells,
    const float* __restrict__ so_p)
{
    __shared__ __align__(16) __bf16 ldsA[2][128 * LDA];
    __shared__ __align__(16) __bf16 ldsB[2][128 * LDA];

    const int m0   = blockIdx.y * 128;
    const int n0   = blockIdx.x * 128;  // over D
    const int t    = threadIdx.x;
    const int lane = t & 31;
    const int wv   = t >> 5;
    const int wr   = wv & 3;    // 4 wave-rows * 32 M
    const int wc   = wv >> 2;   // 2 wave-cols * 64 N
    const int lrow = lane & 15;
    const int kh   = lane >> 4;

    v8f acc[2][4] = {};

    auto issue_tile = [&](int kk, int buf) {
        #pragma unroll
        for (int i = 0; i < 2; ++i) {
            int cc  = t + i * 256;
            int row = cc >> 2, col = (cc & 3) * 8;
            async_b128(lds_off(&ldsA[buf][row * LDA + col]),
                       &hid[(size_t)(m0 + row) * H_DIM + kk + col]);
            async_b128(lds_off(&ldsB[buf][row * LDA + col]),
                       &w2b[(size_t)(n0 + row) * H_DIM + kk + col]);
        }
    };

    auto compute_tile = [&](int buf) {
        Frag a[2], bf[4];
        #pragma unroll
        for (int mi = 0; mi < 2; ++mi) {
            int r = wr * 32 + mi * 16 + lrow;
            a[mi].h[0] = *(const v8bf*)&ldsA[buf][r * LDA + kh * 16];
            a[mi].h[1] = *(const v8bf*)&ldsA[buf][r * LDA + kh * 16 + 8];
        }
        #pragma unroll
        for (int ni = 0; ni < 4; ++ni) {
            int r = wc * 64 + ni * 16 + lrow;
            bf[ni].h[0] = *(const v8bf*)&ldsB[buf][r * LDA + kh * 16];
            bf[ni].h[1] = *(const v8bf*)&ldsB[buf][r * LDA + kh * 16 + 8];
        }
        #pragma unroll
        for (int mi = 0; mi < 2; ++mi)
            #pragma unroll
            for (int ni = 0; ni < 4; ++ni)
                acc[mi][ni] = __builtin_amdgcn_wmma_f32_16x16x32_bf16(
                    false, a[mi].v, false, bf[ni].v, (short)0,
                    acc[mi][ni], false, false);
    };

    issue_tile(0, 0);

    for (int kk = 0; kk < H_DIM - 64; kk += 64) {
        issue_tile(kk + 32, 1);
        wait_async4();
        __syncthreads();
        compute_tile(0);
        __syncthreads();

        issue_tile(kk + 64, 0);
        wait_async4();
        __syncthreads();
        compute_tile(1);
        __syncthreads();
    }
    issue_tile(H_DIM - 32, 1);
    wait_async4();
    __syncthreads();
    compute_tile(0);
    __syncthreads();
    wait_async0();
    __syncthreads();
    compute_tile(1);

    const float s2  = __uint_as_float(cells[2]) * (1.0f / 127.0f);
    const float so  = so_p[0];
    const float iso = 1.0f / so;

    #pragma unroll
    for (int mi = 0; mi < 2; ++mi)
        #pragma unroll
        for (int ni = 0; ni < 4; ++ni) {
            int baser = m0 + wr * 32 + mi * 16 + kh * 8;
            int col   = n0 + wc * 64 + ni * 16 + lrow;
            #pragma unroll
            for (int r = 0; r < 8; ++r) {
                float y  = acc[mi][ni][r] * s2;
                float yq = fminf(fmaxf(rintf(y * iso), -128.0f), 127.0f) * so;
                out[(size_t)(baser + r) * D_DIM + col] = yq;
            }
        }
}

// ---------------------------------------------------------------------------
// Host launcher
// ---------------------------------------------------------------------------
extern "C" void kernel_launch(void* const* d_in, const int* in_sizes, int n_in,
                              void* d_out, int out_size, void* d_ws, size_t ws_size,
                              hipStream_t stream) {
    const float* x  = (const float*)d_in[0];
    const float* w1 = (const float*)d_in[1];
    const float* w3 = (const float*)d_in[2];
    const float* w2 = (const float*)d_in[3];
    const float* sg = (const float*)d_in[4];
    const float* su = (const float*)d_in[5];
    const float* so = (const float*)d_in[6];
    float* out = (float*)d_out;

    char* ws = (char*)d_ws;
    const size_t XB  = (size_t)M_TOT * D_DIM * 2;
    const size_t WB  = (size_t)H_DIM * D_DIM * 2;
    const size_t HB  = (size_t)M_TOT * H_DIM * 2;
    __bf16* xb   = (__bf16*)(ws);
    __bf16* w1b  = (__bf16*)(ws + XB);
    __bf16* w3b  = (__bf16*)(ws + XB + WB);
    __bf16* w2b  = (__bf16*)(ws + XB + 2 * WB);
    __bf16* hid  = (__bf16*)(ws + XB + 3 * WB);
    unsigned int* cells = (unsigned int*)(ws + XB + 3 * WB + HB);

    init_cells_kernel<<<1, 32, 0, stream>>>(cells);

    const size_t nW = (size_t)H_DIM * D_DIM;   // 16.7M each
    absmax_kernel<<<512, 256, 0, stream>>>(w1, nW, cells + 0);
    absmax_kernel<<<512, 256, 0, stream>>>(w3, nW, cells + 1);
    absmax_kernel<<<512, 256, 0, stream>>>(w2, nW, cells + 2);

    quant_weight_kernel<<<1024, 256, 0, stream>>>(w1, w1b, nW, cells + 0);
    quant_weight_kernel<<<1024, 256, 0, stream>>>(w3, w3b, nW, cells + 1);
    quant_weight_kernel<<<1024, 256, 0, stream>>>(w2, w2b, nW, cells + 2);
    cvt_bf16_kernel<<<1024, 256, 0, stream>>>(x, xb, (size_t)M_TOT * D_DIM);

    dim3 g1(H_DIM / 64, M_TOT / 128);          // 128 x 64 blocks
    gemm1_swiglu_kernel<<<g1, 256, 0, stream>>>(xb, w1b, w3b, hid, cells, sg, su);

    dim3 g2(D_DIM / 128, M_TOT / 128);         // 16 x 64 blocks
    gemm2_kernel<<<g2, 256, 0, stream>>>(hid, w2b, out, cells, so);
}
